// LSTM_my_9285719294336
// MI455X (gfx1250) — matrix operand order repaired and checked by reading, not compiled, not run on hardware
//
#include <hip/hip_runtime.h>

// ---------------------------------------------------------------------------
// LSTM for MI455X (gfx1250, wave32, WMMA).  B=32, T=512, I=H=1024, G=4096.
//   1) convert Wi,Wh -> bf16; x -> bf16 time-major; h0 -> bf16; zero barrier
//   2) xproj = x_tm @ Wi^T + bi  (WMMA GEMM, high occupancy hides latency)
//   3) ONE persistent kernel runs all 512 recurrent steps:
//        - 128 blocks (co-resident), each owns (batch-tile, hidden-tile)
//          for all 4 gates; K split over 4 waves + LDS reduce
//        - the block's 128 KB Wh tile is staged into LDS ONCE (CDNA5 has
//          320 KB LDS/WGP) -> steady state B-fragment reads are ds_load,
//          not L2 round trips
//        - all 8 A-fragments of a step hoisted into registers up front
//          (one clause of global loads -> one latency exposure per step)
//        - cell state c held in wave-0 registers for the whole scan
//        - software grid barrier (agent-scope atomics + s_sleep) per step
// ---------------------------------------------------------------------------

typedef __bf16 bf16_t;
typedef __attribute__((ext_vector_type(16))) __bf16 bf16x16;
typedef __attribute__((ext_vector_type(8)))  __bf16 bf16x8;
typedef __attribute__((ext_vector_type(8)))  float  f32x8;

#define LSTM_B 32
#define LSTM_T 512
#define LSTM_I 1024
#define LSTM_H 1024
#define LSTM_G 4096
#define SCAN_NBLK 128u

union ABFrag { bf16x16 v; bf16x8 h[2]; };

// A fragment (16x32 bf16 MxK): lanes 0-15 rows M, k={0..15};
// lanes 16-31 same rows, k={16..31}.  Two contiguous 16B loads.
__device__ __forceinline__ bf16x16 load_a_frag(const bf16_t* A, int lda,
                                               int m0, int kb, int l, int half) {
  const bf16_t* row = A + (size_t)(m0 + l) * lda + kb;
  ABFrag u;
  u.h[0] = *(const bf16x8*)(row + half * 16);
  u.h[1] = *(const bf16x8*)(row + half * 16 + 8);
  return u.v;
}

// B fragment (32x16 bf16 KxN) from row-major W[n][k]: lane (half,l) holds
// n=n0+l, k=kb+half*16+i -> one contiguous 32B read.
__device__ __forceinline__ bf16x16 load_b_frag(const bf16_t* W, int ldw,
                                               int n0, int kb, int l, int half) {
  return *(const bf16x16*)(W + (size_t)(n0 + l) * ldw + kb + half * 16);
}

__device__ __forceinline__ float sigmoidf_(float x) {
  return 1.0f / (1.0f + __expf(-x));
}

// Software grid barrier: all SCAN_NBLK blocks are co-resident by construction.
__device__ __forceinline__ void grid_barrier(unsigned* cnt, unsigned* gen) {
  __syncthreads();
  if (threadIdx.x == 0) {
    __threadfence();  // flush h/out stores to device scope
    unsigned g = __hip_atomic_load(gen, __ATOMIC_RELAXED, __HIP_MEMORY_SCOPE_AGENT);
    unsigned arrived = __hip_atomic_fetch_add(cnt, 1u, __ATOMIC_ACQ_REL,
                                              __HIP_MEMORY_SCOPE_AGENT);
    if (arrived == SCAN_NBLK - 1u) {
      __hip_atomic_store(cnt, 0u, __ATOMIC_RELAXED, __HIP_MEMORY_SCOPE_AGENT);
      __hip_atomic_fetch_add(gen, 1u, __ATOMIC_RELEASE, __HIP_MEMORY_SCOPE_AGENT);
    } else {
      while (__hip_atomic_load(gen, __ATOMIC_RELAXED,
                               __HIP_MEMORY_SCOPE_AGENT) == g)
        __builtin_amdgcn_s_sleep(1);
    }
    __threadfence();  // invalidate stale lines before next step's h reads
  }
  __syncthreads();
}

// ---------------------------------------------------------------------------
// Phase 0: conversions + init
// ---------------------------------------------------------------------------
__global__ __launch_bounds__(256) void lstm_convert_w(
    const float* __restrict__ Wi, const float* __restrict__ Wh,
    bf16_t* __restrict__ Wi_b, bf16_t* __restrict__ Wh_b) {
  size_t e = (size_t)blockIdx.x * 256 + threadIdx.x;  // 4096*1024
  Wi_b[e] = (bf16_t)Wi[e];
  Wh_b[e] = (bf16_t)Wh[e];
}

__global__ __launch_bounds__(256) void lstm_convert_x(
    const float* __restrict__ x, bf16_t* __restrict__ x_tm) {
  size_t e = (size_t)blockIdx.x * 256 + threadIdx.x;  // B*T*I
  int i = (int)(e & (LSTM_I - 1));
  int t = (int)((e >> 10) & (LSTM_T - 1));
  int b = (int)(e >> 19);
  x_tm[((size_t)t * LSTM_B + b) * LSTM_I + i] = (bf16_t)x[e];
}

__global__ __launch_bounds__(256) void lstm_init_state(
    const float* __restrict__ h0, bf16_t* __restrict__ hbuf,
    unsigned* __restrict__ bar) {
  int e = blockIdx.x * 256 + threadIdx.x;  // B*H = 32768
  hbuf[e] = (bf16_t)h0[e];
  if (e == 0) { bar[0] = 0u; bar[1] = 0u; }
}

// ---------------------------------------------------------------------------
// Phase 1: xproj[m=t*B+b][g] = sum_i x_tm[m][i]*Wi[g][i] + bi[g]
// One wave: one 16-row M tile x four 16-col N tiles.  65536 waves of
// occupancy hide the load latency here.
// ---------------------------------------------------------------------------
__global__ __launch_bounds__(128) void lstm_xproj_gemm(
    const bf16_t* __restrict__ x_tm, const bf16_t* __restrict__ Wi_b,
    const float* __restrict__ bi, float* __restrict__ xproj) {
  int wid  = blockIdx.x * 4 + (threadIdx.x >> 5);
  int lane = threadIdx.x & 31;
  int l = lane & 15, half = lane >> 4;
  int mt = wid >> 6;          // 0..1023
  int ng = wid & 63;          // 0..63
  int m0 = mt * 16;
  int n0 = ng * 64;

  f32x8 acc[4];
#pragma unroll
  for (int j = 0; j < 4; ++j) acc[j] = (f32x8){};

#pragma unroll 4
  for (int kb = 0; kb < LSTM_I; kb += 32) {
    bf16x16 a = load_a_frag(x_tm, LSTM_I, m0, kb, l, half);
#pragma unroll
    for (int j = 0; j < 4; ++j) {
      bf16x16 b = load_b_frag(Wi_b, LSTM_I, n0 + j * 16, kb, l, half);
      acc[j] = __builtin_amdgcn_wmma_f32_16x16x32_bf16(
          false, a, false, b, (short)0, acc[j], false, false);
    }
  }

#pragma unroll
  for (int j = 0; j < 4; ++j) {
#pragma unroll
    for (int r = 0; r < 8; ++r) {
      int m = m0 + half * 8 + r;
      int g = n0 + j * 16 + l;
      xproj[(size_t)m * LSTM_G + g] = acc[j][r] + bi[g];
    }
  }
}

// ---------------------------------------------------------------------------
// Phase 2: persistent scan kernel — all 512 timesteps in one launch.
// grid = (H/16=64, B/16=2) = 128 blocks, 128 threads (4 waves) each.
// LDS: 128 KB Wh tile + 16 KB reduction = 144 KB (of 320 KB/WGP).
// ---------------------------------------------------------------------------
__global__ __launch_bounds__(128) void lstm_scan(
    const float* __restrict__ xproj,   // [T*B, 4096]
    const bf16_t* __restrict__ Wh_b,   // [4096, 1024] bf16
    const float* __restrict__ bh,      // [4096]
    const float* __restrict__ c0,      // [32, 1024]
    bf16_t* hbuf0, bf16_t* hbuf1,      // h ping-pong, bf16 [32,1024]
    float* __restrict__ out,           // [B, T, H]
    float* __restrict__ hT, float* __restrict__ cT,
    unsigned* bar) {
  // whs[row=g*16+col][k]: this block's Wh tile, resident in LDS for all steps
  __shared__ bf16_t whs[64][LSTM_H];   // 128 KB
  __shared__ float red[4][4][32][8];   // [wave][gate][lane][r] = 16 KB

  int tid  = threadIdx.x;
  int w    = tid >> 5;
  int lane = tid & 31;
  int l = lane & 15, half = lane >> 4;
  int nt = blockIdx.x;                 // 0..63  (hidden tile)
  int mt = blockIdx.y;                 // 0..1   (batch tile)
  int m0 = mt * 16;
  int k0 = w * 256;
  int hh = nt * 16 + l;

  // ---- stage Wh tile into LDS once: 64 rows x 2 KB, 16 B per thread/row ----
  for (int row = 0; row < 64; ++row) {
    size_t grow = (size_t)((row >> 4) * LSTM_H + nt * 16 + (row & 15));
    ((bf16x8*)&whs[row][0])[tid] =
        ((const bf16x8*)(Wh_b + grow * LSTM_H))[tid];
  }

  // Wave-0 persistent state: cell state + recurrent bias in registers.
  float c_reg[8];
  float bhv[4];
  if (w == 0) {
#pragma unroll
    for (int r = 0; r < 8; ++r)
      c_reg[r] = c0[(size_t)(m0 + half * 8 + r) * LSTM_H + hh];
#pragma unroll
    for (int g = 0; g < 4; ++g) bhv[g] = bh[g * LSTM_H + hh];
  }
  __syncthreads();

#pragma unroll 1
  for (int t = 0; t < LSTM_T; ++t) {
    const bf16_t* hp = (t & 1) ? hbuf1 : hbuf0;
    bf16_t*       hn = (t & 1) ? hbuf0 : hbuf1;

    // Hoist ALL A-fragments for this wave's K slice: 16 clause'd global
    // loads, one latency exposure per step.
    bf16x16 af[8];
#pragma unroll
    for (int it = 0; it < 8; ++it)
      af[it] = load_a_frag(hp, LSTM_H, m0, k0 + it * 32, l, half);

    f32x8 acc[4];
#pragma unroll
    for (int g = 0; g < 4; ++g) acc[g] = (f32x8){};

#pragma unroll
    for (int it = 0; it < 8; ++it) {
      int kb = k0 + it * 32;
#pragma unroll
      for (int g = 0; g < 4; ++g) {
        // B fragment from LDS: one contiguous 32B read per lane
        bf16x16 b = *(const bf16x16*)&whs[g * 16 + l][kb + half * 16];
        acc[g] = __builtin_amdgcn_wmma_f32_16x16x32_bf16(
            false, af[it], false, b, (short)0, acc[g], false, false);
      }
    }

#pragma unroll
    for (int g = 0; g < 4; ++g)
#pragma unroll
      for (int r = 0; r < 8; ++r) red[w][g][lane][r] = acc[g][r];
    __syncthreads();

    if (w == 0) {
      float s[4][8];
#pragma unroll
      for (int g = 0; g < 4; ++g)
#pragma unroll
        for (int r = 0; r < 8; ++r)
          s[g][r] = red[0][g][lane][r] + red[1][g][lane][r] +
                    red[2][g][lane][r] + red[3][g][lane][r];

#pragma unroll
      for (int r = 0; r < 8; ++r) {
        int b = m0 + half * 8 + r;
        size_t xbase = ((size_t)t * LSTM_B + b) * LSTM_G;
        float iv = sigmoidf_(s[0][r] + xproj[xbase + 0 * LSTM_H + hh] + bhv[0]);
        float fv = sigmoidf_(s[1][r] + xproj[xbase + 1 * LSTM_H + hh] + bhv[1]);
        float gv = tanhf   (s[2][r] + xproj[xbase + 2 * LSTM_H + hh] + bhv[2]);
        float ov = sigmoidf_(s[3][r] + xproj[xbase + 3 * LSTM_H + hh] + bhv[3]);
        float c_new = fv * c_reg[r] + iv * gv;
        float h_new = ov * tanhf(c_new);
        c_reg[r] = c_new;
        size_t ci = (size_t)b * LSTM_H + hh;
        hn[ci] = (bf16_t)h_new;
        out[((size_t)b * LSTM_T + t) * LSTM_H + hh] = h_new;
        if (t == LSTM_T - 1) { hT[ci] = h_new; cT[ci] = c_new; }
      }
    }

    grid_barrier(&bar[0], &bar[1]);
  }
}

// ---------------------------------------------------------------------------
// Host launcher
// ---------------------------------------------------------------------------
extern "C" void kernel_launch(void* const* d_in, const int* in_sizes, int n_in,
                              void* d_out, int out_size, void* d_ws,
                              size_t ws_size, hipStream_t stream) {
  (void)in_sizes; (void)n_in; (void)out_size; (void)ws_size;
  const float* x  = (const float*)d_in[0];   // [B,T,I]
  const float* h0 = (const float*)d_in[1];   // [B,H]
  const float* c0 = (const float*)d_in[2];   // [B,H]
  const float* Wi = (const float*)d_in[3];   // [4H,I]
  const float* bi = (const float*)d_in[4];   // [4H]
  const float* Wh = (const float*)d_in[5];   // [4H,H]
  const float* bh = (const float*)d_in[6];   // [4H]

  float* out = (float*)d_out;                          // [B,T,H]
  float* hT  = out + (size_t)LSTM_B * LSTM_T * LSTM_H;
  float* cT  = hT + (size_t)LSTM_B * LSTM_H;

  char* ws = (char*)d_ws;
  const size_t XPROJ_BYTES = (size_t)LSTM_T * LSTM_B * LSTM_G * 4;  // 256 MB
  const size_t W_BYTES     = (size_t)LSTM_G * LSTM_I * 2;           // 8 MB
  const size_t XTM_BYTES   = (size_t)LSTM_T * LSTM_B * LSTM_I * 2;  // 32 MB
  const size_t HBUF_BYTES  = (size_t)LSTM_B * LSTM_H * 2;           // 64 KB

  float*    xproj = (float*)ws;
  bf16_t*   Wi_b  = (bf16_t*)(ws + XPROJ_BYTES);
  bf16_t*   Wh_b  = (bf16_t*)(ws + XPROJ_BYTES + W_BYTES);
  bf16_t*   x_tm  = (bf16_t*)(ws + XPROJ_BYTES + 2 * W_BYTES);
  bf16_t*   hbuf0 = (bf16_t*)(ws + XPROJ_BYTES + 2 * W_BYTES + XTM_BYTES);
  bf16_t*   hbuf1 = (bf16_t*)((char*)hbuf0 + HBUF_BYTES);
  unsigned* bar   = (unsigned*)((char*)hbuf1 + HBUF_BYTES);

  // Phase 0
  lstm_convert_w<<<(LSTM_G * LSTM_I) / 256, 256, 0, stream>>>(Wi, Wh, Wi_b, Wh_b);
  lstm_convert_x<<<(LSTM_B * LSTM_T * LSTM_I) / 256, 256, 0, stream>>>(x, x_tm);
  lstm_init_state<<<(LSTM_B * LSTM_H) / 256, 256, 0, stream>>>(h0, hbuf0, bar);

  // Phase 1: batched input projection
  lstm_xproj_gemm<<<16384, 128, 0, stream>>>(x_tm, Wi_b, bi, xproj);

  // Phase 2: entire recurrence in one persistent kernel (128 blocks, all
  // co-resident; Wh tile pinned in LDS; software grid barrier per timestep).
  lstm_scan<<<dim3(LSTM_H / 16, LSTM_B / 16), 128, 0, stream>>>(
      xproj, Wh_b, bh, c0, hbuf0, hbuf1, out, hT, cT, bar);
}
